// MultiHeadAttention_63385127355136
// MI455X (gfx1250) — matrix operand order repaired and checked by reading, not compiled, not run on hardware
//
#include <hip/hip_runtime.h>
#include <hip/hip_bf16.h>
#include <math.h>

// ---------------------------------------------------------------------------
// MultiHeadAttention forward for MI455X (gfx1250, wave32, WMMA + async-LDS).
//   N=8, S=1024, E=1024, H=16, DK=64
// Pipeline:
//   1) gemm_bias<f32 A, f16 out> x3  -> Q,K,V (f16, [8192][1024]) in ws
//   2) flash-attention (WMMA scores + WMMA PV, online softmax,
//      K tiles DMA'd to LDS with global_load_async_to_lds_b128,
//      double-buffered)                                        -> ctx f16
//   3) gemm_bias<f16 A, f32 out>     -> d_out = ctx @ Wo + bo
// Workspace requirement: 4 * 8192*1024 * 2B = 64 MiB.
// ---------------------------------------------------------------------------

#define MHA_N 8
#define MHA_S 1024
#define MHA_E 1024
#define MHA_H 16
#define MHA_DK 64
#define MHA_M (MHA_N * MHA_S)   // 8192 rows

typedef _Float16 half4  __attribute__((ext_vector_type(4)));
typedef _Float16 half8  __attribute__((ext_vector_type(8)));
typedef _Float16 v16h   __attribute__((ext_vector_type(16)));
typedef float    v8f    __attribute__((ext_vector_type(8)));

// ---------------- async copy to LDS (CDNA5 ASYNCcnt path) ------------------
#if defined(__has_builtin)
#if __has_builtin(__builtin_amdgcn_global_load_async_to_lds_b128)
#define HAVE_ASYNC_LDS 1
#endif
#endif
#ifndef HAVE_ASYNC_LDS
#define HAVE_ASYNC_LDS 0
#endif

// Builtin prototype uses GCC-vector int4 pointees in AS(1)/AS(3).
typedef int v4i_b __attribute__((vector_size(16)));
typedef __attribute__((address_space(1))) v4i_b gv4i_t;
typedef __attribute__((address_space(3))) v4i_b lv4i_t;

// Copy 16 bytes global -> LDS per lane. Async (fire-and-forget, ASYNCcnt) when
// the gfx1250 builtin is available; plain load/store fallback otherwise.
static __device__ __forceinline__ void async_b128(const void* g, void* l) {
#if HAVE_ASYNC_LDS
    __builtin_amdgcn_global_load_async_to_lds_b128((gv4i_t*)g, (lv4i_t*)l, 0, 0);
#else
    *(half8*)l = *(const half8*)g;
#endif
}

static __device__ __forceinline__ void async_wait_all() {
#if HAVE_ASYNC_LDS
    asm volatile("s_wait_asynccnt 0x0" ::: "memory");
#endif
}

// ---------------------------- WMMA helpers ---------------------------------
union HV { v16h v; half8 h[2]; };

static __device__ __forceinline__ v16h mkfrag(half8 lo, half8 hi) {
    HV u; u.h[0] = lo; u.h[1] = hi; return u.v;
}

static __device__ __forceinline__ v8f wmma_f16(v16h a, v16h b, v8f c) {
    // D(16x16 f32) = A(16x32 f16) * B(32x16 f16) + C
    return __builtin_amdgcn_wmma_f32_16x16x32_f16(
        /*neg_a=*/false, a, /*neg_b=*/false, b,
        /*c_mod=*/(short)0, c, /*reuse_a=*/false, /*reuse_b=*/false);
}

// A-fragment (16x32, 16-bit): lane row = L%16, g = L/16.
// halves e=0..15 -> K = 16g + e
static __device__ __forceinline__ v16h frag_a(const _Float16* rowbase, int g) {
    half8 lo = *(const half8*)(rowbase + 16 * g);
    half8 hi = *(const half8*)(rowbase + 16 * g + 8);
    return mkfrag(lo, hi);
}

// B-fragment (32x16, 16-bit): lane col = L%16, g = L/16, halves e -> K = 16g + e.
// colbase points at the lane's column laid out contiguously in K.
static __device__ __forceinline__ v16h frag_b(const _Float16* colbase, int g) {
    half8 lo = *(const half8*)(colbase + 16 * g);
    half8 hi = *(const half8*)(colbase + 16 * g + 8);
    return mkfrag(lo, hi);
}

// ---------------------------------------------------------------------------
// Tile staging for the GEMM: A tile 128x32 (async when f16 source), W tile
// 32x128 converted f32->f16 and transposed into sB[n][k].
// ---------------------------------------------------------------------------
template <bool A_HALF>
static __device__ __forceinline__ void stage_gemm_tiles(
    _Float16 (*sA)[32], _Float16 (*sB)[32],
    const void* __restrict__ Ain, const float* __restrict__ W,
    int bm, int bn, int kk, int K, int Nn, int tid) {
    if (A_HALF) {
        const _Float16* Ap = (const _Float16*)Ain;
#pragma unroll
        for (int i = 0; i < 2; ++i) {            // 512 x 16B chunks, 2/thread
            const int idx = tid + 256 * i;
            const int row = idx >> 2;            // 0..127
            const int seg = (idx & 3) * 8;       // half offset, 16B aligned
            async_b128(Ap + (size_t)(bm + row) * K + kk + seg, &sA[row][seg]);
        }
    } else {
#pragma unroll
        for (int i = 0; i < 4; ++i) {
            const int row = (tid >> 3) + 32 * i;
            const int col = (tid & 7) * 4;
            const float* Af = (const float*)Ain;
            float4 f = *(const float4*)(Af + (size_t)(bm + row) * K + kk + col);
            half4 h; h[0] = (_Float16)f.x; h[1] = (_Float16)f.y;
            h[2] = (_Float16)f.z; h[3] = (_Float16)f.w;
            *(half4*)&sA[row][col] = h;
        }
    }
#pragma unroll
    for (int i = 0; i < 4; ++i) {
        const int krow = (tid >> 5) + 8 * i;
        const int col4 = (tid & 31) * 4;
        float4 f = *(const float4*)(W + (size_t)(kk + krow) * Nn + bn + col4);
        sB[col4 + 0][krow] = (_Float16)f.x;
        sB[col4 + 1][krow] = (_Float16)f.y;
        sB[col4 + 2][krow] = (_Float16)f.z;
        sB[col4 + 3][krow] = (_Float16)f.w;
    }
}

// ---------------------------------------------------------------------------
// Stage 1 & 3: double-buffered tiled WMMA GEMM with bias.
// Out[M,Nn] = A[M,K] * W[K,Nn] + bias.  Block 256 threads (8 waves),
// tile 128x128, K-step 32; wave grid 4 (row) x 2 (col), 2x4 accum tiles.
// ---------------------------------------------------------------------------
template <bool A_HALF, bool OUT_HALF>
__global__ __launch_bounds__(256)
void gemm_bias_wmma(const void* __restrict__ Ain,
                    const float* __restrict__ W,
                    const float* __restrict__ bias,
                    void* __restrict__ Out,
                    int M, int K, int Nn) {
    __shared__ _Float16 sA[2][128][32];   // [buf][m][k]  16 KB
    __shared__ _Float16 sB[2][128][32];   // [buf][n][k]  16 KB

    const int tid  = threadIdx.x;
    const int lane = tid & 31;
    const int w    = tid >> 5;
    const int ln   = lane & 15;
    const int g    = lane >> 4;
    const int wr   = w & 3;        // row group: 32 rows
    const int wc   = w >> 2;       // col group: 64 cols
    const int bm   = blockIdx.y * 128;
    const int bn   = blockIdx.x * 128;

    v8f acc[2][4];
#pragma unroll
    for (int rt = 0; rt < 2; ++rt)
#pragma unroll
        for (int ct = 0; ct < 4; ++ct)
#pragma unroll
            for (int r = 0; r < 8; ++r) acc[rt][ct][r] = 0.0f;

    const int nk = K / 32;
    stage_gemm_tiles<A_HALF>(sA[0], sB[0], Ain, W, bm, bn, 0, K, Nn, tid);
    async_wait_all();
    __syncthreads();

    for (int it = 0; it < nk; ++it) {
        const int cur = it & 1;
        if (it + 1 < nk)   // prefetch next tile into the other buffer
            stage_gemm_tiles<A_HALF>(sA[cur ^ 1], sB[cur ^ 1], Ain, W,
                                     bm, bn, (it + 1) * 32, K, Nn, tid);

        v16h a[2], b[4];
#pragma unroll
        for (int rt = 0; rt < 2; ++rt)
            a[rt] = frag_a(&sA[cur][wr * 32 + rt * 16 + ln][0], g);
#pragma unroll
        for (int ct = 0; ct < 4; ++ct)
            b[ct] = frag_b(&sB[cur][wc * 64 + ct * 16 + ln][0], g);
#pragma unroll
        for (int rt = 0; rt < 2; ++rt)
#pragma unroll
            for (int ct = 0; ct < 4; ++ct)
                acc[rt][ct] = wmma_f16(a[rt], b[ct], acc[rt][ct]);

        async_wait_all();
        __syncthreads();
    }

    // ---- epilogue: bias + store (C layout: row = r + 8g, col = lane%16) ----
#pragma unroll
    for (int ct = 0; ct < 4; ++ct) {
        const int col = bn + wc * 64 + ct * 16 + ln;
        const float bv = bias[col];
#pragma unroll
        for (int rt = 0; rt < 2; ++rt) {
#pragma unroll
            for (int r = 0; r < 8; ++r) {
                const int row = bm + wr * 32 + rt * 16 + r + 8 * g;
                const float v = acc[rt][ct][r] + bv;
                if (OUT_HALF)
                    ((_Float16*)Out)[(size_t)row * Nn + col] = (_Float16)v;
                else
                    ((float*)Out)[(size_t)row * Nn + col] = v;
            }
        }
    }
}

// ---------------------------------------------------------------------------
// Stage 2: flash attention per (n,h). Block = 256 threads (8 waves), each wave
// owns 16 query rows (block covers 128). Streams keys in tiles of 32 with a
// double-buffered LDS pipeline; K tiles via async DMA to LDS.
// ---------------------------------------------------------------------------
__global__ __launch_bounds__(256)
void flash_attn_wmma(const _Float16* __restrict__ Q,
                     const _Float16* __restrict__ Km,
                     const _Float16* __restrict__ V,
                     const int* __restrict__ mask,
                     _Float16* __restrict__ ctx) {
    __shared__ _Float16 sK[2][32][64];     // [buf][key][d]  row-major, 8 KB
    __shared__ _Float16 sV[2][64][32];     // [buf][d][key]  transposed, 8 KB
    __shared__ _Float16 sP[8][16][32];     // per-wave P tile (f16), 8 KB

    const int tid  = threadIdx.x;
    const int lane = tid & 31;
    const int w    = tid >> 5;
    const int ln   = lane & 15;
    const int g    = lane >> 4;

    const int nh = blockIdx.y;
    const int n  = nh / MHA_H;
    const int h  = nh % MHA_H;
    const size_t headoff = (size_t)n * MHA_S * MHA_E + (size_t)h * MHA_DK;
    const int qrow0 = blockIdx.x * 128 + w * 16;

    // Preload Q A-fragments straight from global (two b128 loads each).
    const _Float16* Qrow = Q + headoff + (size_t)(qrow0 + ln) * MHA_E;
    v16h qa[2];
#pragma unroll
    for (int j = 0; j < 2; ++j) qa[j] = frag_a(Qrow + 32 * j, g);

    v8f o[4];
#pragma unroll
    for (int ct = 0; ct < 4; ++ct)
#pragma unroll
        for (int r = 0; r < 8; ++r) o[ct][r] = 0.0f;

    float m[8], l[8];
#pragma unroll
    for (int r = 0; r < 8; ++r) { m[r] = -INFINITY; l[r] = 0.0f; }

    const float scale = 0.125f;  // 1/sqrt(64)
    const int* maskn = mask + (size_t)n * MHA_S * MHA_S;

    // Stage one 32-key tile: K via async DMA, V via manual f16 transpose.
    auto stage_kv = [&](int buf, int k0) {
        const int key = tid >> 3;            // 0..31
        const int dg  = (tid & 7) * 8;       // 0..56
        async_b128(Km + headoff + (size_t)(k0 + key) * MHA_E + dg,
                   &sK[buf][key][dg]);
        half8 vv = *(const half8*)(V + headoff + (size_t)(k0 + key) * MHA_E + dg);
#pragma unroll
        for (int jj = 0; jj < 8; ++jj) sV[buf][dg + jj][key] = vv[jj];
    };

    const int nkt = MHA_S / 32;
    stage_kv(0, 0);
    async_wait_all();
    __syncthreads();

    for (int it = 0; it < nkt; ++it) {
        const int cur = it & 1;
        const int k0  = it * 32;
        if (it + 1 < nkt) stage_kv(cur ^ 1, k0 + 32);   // prefetch next tile

        // ---- load all four K B-fragments up front (distinct registers so the
        //      8 ds_load_b128 issue together and WMMAs use partial waits) ----
        v16h kb[2][2];
#pragma unroll
        for (int t = 0; t < 2; ++t) {
            const int keyl = 16 * t + ln;            // key within tile
#pragma unroll
            for (int j = 0; j < 2; ++j)
                kb[t][j] = mkfrag(*(const half8*)&sK[cur][keyl][32 * j + 16 * g],
                                  *(const half8*)&sK[cur][keyl][32 * j + 16 * g + 8]);
        }

        // ---- scores: S = Q * K^T for 32 keys (2 column tiles, 2 k-steps) ----
        v8f s[2];
#pragma unroll
        for (int t = 0; t < 2; ++t) {
            v8f a;
#pragma unroll
            for (int r = 0; r < 8; ++r) a[r] = 0.0f;
#pragma unroll
            for (int j = 0; j < 2; ++j)
                a = wmma_f16(qa[j], kb[t][j], a);
            // scale + mask (mask!=0 -> -1e9)
            const int* mp = maskn + (k0 + 16 * t + ln);
#pragma unroll
            for (int r = 0; r < 8; ++r) {
                const int q = qrow0 + r + 8 * g;
                const int mv = mp[(size_t)q * MHA_S];
                a[r] = mv ? -1.0e9f : (a[r] * scale);
            }
            s[t] = a;
        }

        // ---- online softmax (row reductions across 16-lane halves) ----
        float alpha[8];
#pragma unroll
        for (int r = 0; r < 8; ++r) {
            float tmax = fmaxf(s[0][r], s[1][r]);
#pragma unroll
            for (int off = 1; off < 16; off <<= 1)
                tmax = fmaxf(tmax, __shfl_xor(tmax, off, 32));
            const float mnew = fmaxf(m[r], tmax);
            alpha[r] = __expf(m[r] - mnew);
            m[r] = mnew;
            s[0][r] = __expf(s[0][r] - mnew);
            s[1][r] = __expf(s[1][r] - mnew);
            float rsum = s[0][r] + s[1][r];
#pragma unroll
            for (int off = 1; off < 16; off <<= 1)
                rsum += __shfl_xor(rsum, off, 32);
            l[r] = l[r] * alpha[r] + rsum;
        }
#pragma unroll
        for (int ct = 0; ct < 4; ++ct)
#pragma unroll
            for (int r = 0; r < 8; ++r) o[ct][r] *= alpha[r];

        // ---- P (16x32) to per-wave LDS as f16 ----
#pragma unroll
        for (int t = 0; t < 2; ++t)
#pragma unroll
            for (int r = 0; r < 8; ++r)
                sP[w][r + 8 * g][16 * t + ln] = (_Float16)s[t][r];

        // V B-fragments: independent of sP, issue before the fence so their
        // latency drains together with the P stores.
        v16h vb[4];
#pragma unroll
        for (int ct = 0; ct < 4; ++ct)
            vb[ct] = frag_b(&sV[cur][16 * ct + ln][0], g);

        // Same-wave LDS ops are in-order; wait drains them and fences the
        // compiler before the A-layout reload of P.
        asm volatile("s_wait_dscnt 0" ::: "memory");
        const v16h pa = frag_a(&sP[w][ln][0], g);

        // ---- O += P * V ----
#pragma unroll
        for (int ct = 0; ct < 4; ++ct)
            o[ct] = wmma_f16(pa, vb[ct], o[ct]);

        async_wait_all();   // our prefetch into buf^1 has landed
        __syncthreads();    // everyone done staging & reading this buffer
    }

    // ---- normalize and store context (f16, [8192][1024]) ----
    _Float16* cp = ctx + headoff;
#pragma unroll
    for (int r = 0; r < 8; ++r) l[r] = 1.0f / l[r];
#pragma unroll
    for (int ct = 0; ct < 4; ++ct) {
        const int col = 16 * ct + ln;
#pragma unroll
        for (int r = 0; r < 8; ++r) {
            const int row = qrow0 + r + 8 * g;
            cp[(size_t)row * MHA_E + col] = (_Float16)(o[ct][r] * l[r]);
        }
    }
}

// ---------------------------------------------------------------------------
// Host launcher
// ---------------------------------------------------------------------------
extern "C" void kernel_launch(void* const* d_in, const int* in_sizes, int n_in,
                              void* d_out, int out_size, void* d_ws, size_t ws_size,
                              hipStream_t stream) {
    (void)in_sizes; (void)n_in; (void)out_size; (void)ws_size;
    const float* x    = (const float*)d_in[0];
    const int*   mask = (const int*)d_in[1];   // bool mask as int32
    const float* Wq   = (const float*)d_in[2];
    const float* bq   = (const float*)d_in[3];
    const float* Wk   = (const float*)d_in[4];
    const float* bk   = (const float*)d_in[5];
    const float* Wv   = (const float*)d_in[6];
    const float* bv   = (const float*)d_in[7];
    const float* Wo   = (const float*)d_in[8];
    const float* bo   = (const float*)d_in[9];
    float* out = (float*)d_out;

    const size_t plane = (size_t)MHA_M * MHA_E;      // 8192*1024 elements
    _Float16* Qw  = (_Float16*)d_ws;
    _Float16* Kw  = Qw + plane;
    _Float16* Vw  = Kw + plane;
    _Float16* Ctx = Vw + plane;                      // total 64 MiB f16

    const dim3 gg(MHA_E / 128, MHA_M / 128);         // (8, 64)
    gemm_bias_wmma<false, true><<<gg, 256, 0, stream>>>(x, Wq, bq, Qw, MHA_M, MHA_E, MHA_E);
    gemm_bias_wmma<false, true><<<gg, 256, 0, stream>>>(x, Wk, bk, Kw, MHA_M, MHA_E, MHA_E);
    gemm_bias_wmma<false, true><<<gg, 256, 0, stream>>>(x, Wv, bv, Vw, MHA_M, MHA_E, MHA_E);

    const dim3 ga(MHA_S / 128, MHA_N * MHA_H);       // (8, 128)
    flash_attn_wmma<<<ga, 256, 0, stream>>>(Qw, Kw, Vw, mask, Ctx);

    gemm_bias_wmma<true, false><<<gg, 256, 0, stream>>>(Ctx, Wo, bo, out, MHA_M, MHA_E, MHA_E);
}
